// TimeAwareAttention_7035156430998
// MI455X (gfx1250) — matrix operand order repaired and checked by reading, not compiled, not run on hardware
//
#include <hip/hip_runtime.h>

#define B_   4
#define N_   2048
#define DIM_ 512
#define T_   16
#define H_   8
#define DH_  64
// patches per timepoint = N_/T_ = 128

typedef __attribute__((ext_vector_type(16))) __bf16 v16bf;
typedef __attribute__((ext_vector_type(8)))  float  v8f;
typedef __attribute__((ext_vector_type(4)))  unsigned int ui4;

union FragAB { v16bf v; ui4 u[2]; };

__device__ __forceinline__ unsigned short f2bf(float f) {
    unsigned int u = __float_as_uint(f);
    u += 0x7FFFu + ((u >> 16) & 1u);          // round-to-nearest-even
    return (unsigned short)(u >> 16);
}

__device__ __forceinline__ v8f wmma_bf16(const FragAB& a, const FragAB& b, v8f c) {
    return __builtin_amdgcn_wmma_f32_16x16x32_bf16(
        false, a.v, false, b.v, (short)0, c, false, false);
}

__device__ __forceinline__ unsigned lds_off32(const void* p) {
    // generic pointer to __shared__ = {LDS aperture, offset-in-allocation}
    return (unsigned)(unsigned long long)(uintptr_t)p;
}

// async 16B copy global -> LDS (ASYNCcnt-tracked, no VGPR staging)
__device__ __forceinline__ void async_ld16(unsigned lds_dst, const void* gsrc) {
    asm volatile("global_load_async_to_lds_b128 %0, %1, off"
                 :: "v"(lds_dst), "v"(gsrc) : "memory");
}

// ---------------------------------------------------------------------------
// 1) LayerNorm: x[8192,512] fp32 -> xn bf16. One wave per row.
// ---------------------------------------------------------------------------
__global__ void ln_kernel(const float* __restrict__ x,
                          const float* __restrict__ gamma,
                          const float* __restrict__ beta,
                          unsigned short* __restrict__ xn) {
    int wid  = threadIdx.x >> 5;
    int lane = threadIdx.x & 31;
    int row  = blockIdx.x * 4 + wid;          // 8192 rows

    const float4* xp = (const float4*)(x + (size_t)row * DIM_ + lane * 16);
    float v[16];
#pragma unroll
    for (int i = 0; i < 4; i++) {
        float4 f = xp[i];
        v[4*i+0] = f.x; v[4*i+1] = f.y; v[4*i+2] = f.z; v[4*i+3] = f.w;
    }
    float s = 0.f, ss = 0.f;
#pragma unroll
    for (int i = 0; i < 16; i++) { s += v[i]; ss += v[i] * v[i]; }
#pragma unroll
    for (int off = 1; off < 32; off <<= 1) {
        s  += __shfl_xor(s,  off, 32);
        ss += __shfl_xor(ss, off, 32);
    }
    float mean = s * (1.f / DIM_);
    float var  = ss * (1.f / DIM_) - mean * mean;
    float rstd = rsqrtf(var + 1e-5f);

    const float4* gp = (const float4*)(gamma + lane * 16);
    const float4* bp = (const float4*)(beta  + lane * 16);
    float g[16], bt[16];
#pragma unroll
    for (int i = 0; i < 4; i++) {
        float4 fg = gp[i], fb = bp[i];
        g[4*i+0]=fg.x; g[4*i+1]=fg.y; g[4*i+2]=fg.z; g[4*i+3]=fg.w;
        bt[4*i+0]=fb.x; bt[4*i+1]=fb.y; bt[4*i+2]=fb.z; bt[4*i+3]=fb.w;
    }
    unsigned int pk[8];
#pragma unroll
    for (int i = 0; i < 8; i++) {
        float o0 = (v[2*i+0] - mean) * rstd * g[2*i+0] + bt[2*i+0];
        float o1 = (v[2*i+1] - mean) * rstd * g[2*i+1] + bt[2*i+1];
        pk[i] = (unsigned int)f2bf(o0) | ((unsigned int)f2bf(o1) << 16);
    }
    ui4* dst = (ui4*)(xn + (size_t)row * DIM_ + lane * 16);
    ui4 t0, t1;
    t0[0]=pk[0]; t0[1]=pk[1]; t0[2]=pk[2]; t0[3]=pk[3];
    t1[0]=pk[4]; t1[1]=pk[5]; t1[2]=pk[6]; t1[3]=pk[7];
    dst[0] = t0; dst[1] = t1;
}

// ---------------------------------------------------------------------------
// 2) Weight prep: transpose + fp32->bf16 so B-fragments are K-contiguous.
// ---------------------------------------------------------------------------
__global__ void prep_wqkv(const float* __restrict__ W, unsigned short* __restrict__ Wt) {
    int idx = blockIdx.x * 256 + threadIdx.x;         // 1536*512
    int kk = idx & 511, j = idx >> 9;                 // Wt[j][k] = W[k][j]
    Wt[idx] = f2bf(W[(size_t)kk * 1536 + j]);
}
__global__ void prep_wout(const float* __restrict__ W, unsigned short* __restrict__ Wt) {
    int idx = blockIdx.x * 256 + threadIdx.x;         // 512*512
    int kk = idx & 511, j = idx >> 9;
    Wt[idx] = f2bf(W[(size_t)kk * 512 + j]);
}
__global__ void prep_decay(const float* __restrict__ R, const float* __restrict__ a,
                           const float* __restrict__ c, float* __restrict__ dec) {
    int idx = blockIdx.x * 256 + threadIdx.x;         // B*H*T*T = 8192
    int tj = idx & 15, ti = (idx >> 4) & 15, h = (idx >> 8) & 7, b = idx >> 11;
    float r = R[((size_t)b * T_ + ti) * T_ + tj];
    dec[idx] = 1.f / (1.f + __expf(fabsf(a[h]) * r - fabsf(c[h])));
}

// ---------------------------------------------------------------------------
// 3) QKV GEMM: [8192,512]bf16 x [512,1536] -> q/k/vT bf16. 1 wave / 16x16 tile.
// ---------------------------------------------------------------------------
__global__ void gemm_qkv(const unsigned short* __restrict__ xn,
                         const unsigned short* __restrict__ Wt,
                         unsigned short* __restrict__ q,
                         unsigned short* __restrict__ k,
                         unsigned short* __restrict__ vT) {
    int wid = threadIdx.x >> 5, lane = threadIdx.x & 31;
    int tile = blockIdx.x * 4 + wid;                  // 512*96 tiles
    int rt = tile / 96, ct = tile % 96;
    int m0 = rt * 16, j0 = ct * 16;
    int lrow = lane & 15, lhi = lane >> 4;

    const unsigned short* aB = xn + (size_t)(m0 + lrow) * DIM_ + lhi * 16;
    const unsigned short* bB = Wt + (size_t)(j0 + lrow) * DIM_ + lhi * 16;

    v8f acc = {};
#pragma unroll 4
    for (int k0 = 0; k0 < DIM_; k0 += 32) {
        FragAB fa, fb;
        fa.u[0] = *(const ui4*)(aB + k0);
        fa.u[1] = *(const ui4*)(aB + k0 + 8);
        fb.u[0] = *(const ui4*)(bB + k0);
        fb.u[1] = *(const ui4*)(bB + k0 + 8);
        acc = wmma_bf16(fa, fb, acc);
    }
    int j   = j0 + lrow;          // 0..1535
    int sec = j / 512;            // 0=q 1=k 2=v
    int jj  = j & 511;
    int h   = jj >> 6, dh = jj & 63;
#pragma unroll
    for (int r = 0; r < 8; r++) {
        int m  = m0 + r + 8 * lhi;
        int b  = m >> 11, n = m & 2047;
        unsigned short val = f2bf(acc[r]);
        if (sec == 0)
            q[((size_t)((b * H_ + h) * N_ + n)) * DH_ + dh] = val;
        else if (sec == 1)
            k[((size_t)((b * H_ + h) * N_ + n)) * DH_ + dh] = val;
        else
            vT[((size_t)((b * H_ + h) * DH_ + dh)) * N_ + n] = val;
    }
}

// ---------------------------------------------------------------------------
// 4) Flash attention: 4 waves/block share (b,h); K/V chunks are double-buffer
//    staged into LDS with global_load_async_to_lds_b128 (ASYNCcnt), fragments
//    consumed via ds_load_b128; fp32 online softmax; bf16 WMMA GEMMs.
// ---------------------------------------------------------------------------
__global__ void attn_kernel(const unsigned short* __restrict__ q,
                            const unsigned short* __restrict__ k,
                            const unsigned short* __restrict__ vT,
                            const float* __restrict__ dec,
                            unsigned short* __restrict__ ao) {
    __shared__ __align__(16) unsigned short lds_p[4][16 * 32];
    __shared__ __align__(16) unsigned short kbuf[2][32 * 64];   // [key][dh]
    __shared__ __align__(16) unsigned short vbuf[2][64 * 32];   // [dh][key]

    int tid = threadIdx.x;
    int wid = tid >> 5, lane = tid & 31;
    int id = blockIdx.x * 4 + wid;                    // 4096 waves
    int qt = id & 127, h = (id >> 7) & 7, b = id >> 10;
    int n0 = qt * 16, ti = n0 >> 7;
    int lrow = lane & 15, lhi = lane >> 4;

    const unsigned short* qb = q  + ((size_t)(b * H_ + h) * N_)  * DH_;
    const unsigned short* kb = k  + ((size_t)(b * H_ + h) * N_)  * DH_;
    const unsigned short* vb = vT + ((size_t)(b * H_ + h) * DH_) * N_;
    const float* db = dec + ((size_t)(b * H_ + h) * T_ + ti) * T_;

    unsigned kldsb[2] = { lds_off32(&kbuf[0][0]), lds_off32(&kbuf[1][0]) };
    unsigned vldsb[2] = { lds_off32(&vbuf[0][0]), lds_off32(&vbuf[1][0]) };

    FragAB qa[2];
    {
        const unsigned short* qp = qb + (size_t)(n0 + lrow) * DH_ + lhi * 16;
        qa[0].u[0] = *(const ui4*)(qp +  0);
        qa[0].u[1] = *(const ui4*)(qp +  8);
        qa[1].u[0] = *(const ui4*)(qp + 32);
        qa[1].u[1] = *(const ui4*)(qp + 40);
    }
    float rmax[8], rsum[8];
    v8f o[4] = {v8f{}, v8f{}, v8f{}, v8f{}};
#pragma unroll
    for (int r = 0; r < 8; r++) { rmax[r] = 0.f; rsum[r] = 0.f; }

    const float scale = 0.125f;                       // DH^-0.5

    // --- cooperative async stage of one 32-key chunk (4 x 16B per thread) ---
    auto stage = [&](int bsel, int m0s) {
        // K: contiguous 4KB = 256 x 16B pieces; piece p -> lds off p*16
        const unsigned short* ks = kb + (size_t)m0s * DH_ + (size_t)tid * 8;
        async_ld16(kldsb[bsel] + tid * 16,             ks);
        async_ld16(kldsb[bsel] + (tid + 128) * 16,     ks + 128 * 8);
        // V: 64 rows x 64B (strided N_*2 in global); piece p -> lds off p*16
        int p1 = tid + 128;
        const unsigned short* vs0 = vb + (size_t)(tid >> 2) * N_ + m0s + (tid & 3) * 8;
        const unsigned short* vs1 = vb + (size_t)(p1  >> 2) * N_ + m0s + (p1  & 3) * 8;
        async_ld16(vldsb[bsel] + tid * 16, vs0);
        async_ld16(vldsb[bsel] + p1  * 16, vs1);
    };

    stage(0, 0);
    for (int it = 0; it < 64; it++) {
        int cur = it & 1;
        int m0  = it * 32;
        asm volatile("s_wait_asynccnt 0x0" ::: "memory");  // my stage(it) landed
        __syncthreads();                                   // all stages visible
        if (it < 63) stage(cur ^ 1, m0 + 32);              // prefetch next chunk

        float d = db[m0 >> 7];                             // per-chunk decay
        const unsigned short* kB = &kbuf[cur][0];
        const unsigned short* vB = &vbuf[cur][0];

        v8f s[2];
#pragma unroll
        for (int t2 = 0; t2 < 2; t2++) {
            int kr = (t2 * 16 + lrow) * 64 + lhi * 16;
            FragAB k0f, k1f;
            k0f.u[0] = *(const ui4*)(kB + kr);       k0f.u[1] = *(const ui4*)(kB + kr + 8);
            k1f.u[0] = *(const ui4*)(kB + kr + 32);  k1f.u[1] = *(const ui4*)(kB + kr + 40);
            v8f cacc = {};
            cacc = wmma_bf16(qa[0], k0f, cacc);
            cacc = wmma_bf16(qa[1], k1f, cacc);
            s[t2] = cacc;
        }
        float rm_new[8], psum[8];
#pragma unroll
        for (int r = 0; r < 8; r++) {
            float s0 = fmaxf(s[0][r] * scale, 0.f) * d;
            float s1 = fmaxf(s[1][r] * scale, 0.f) * d;
            float m = fmaxf(s0, s1);
#pragma unroll
            for (int off = 1; off < 16; off <<= 1) m = fmaxf(m, __shfl_xor(m, off, 16));
            float mn = fmaxf(rmax[r], m);
            rm_new[r] = mn;
            float p0 = __expf(s0 - mn);
            float p1 = __expf(s1 - mn);
            int rr = (r + 8 * lhi) * 32;
            lds_p[wid][rr + lrow]      = f2bf(p0);
            lds_p[wid][rr + 16 + lrow] = f2bf(p1);
            float ps = p0 + p1;
#pragma unroll
            for (int off = 1; off < 16; off <<= 1) ps += __shfl_xor(ps, off, 16);
            psum[r] = ps;
        }
#pragma unroll
        for (int r = 0; r < 8; r++) {
            float cf = __expf(rmax[r] - rm_new[r]);
            rsum[r] = rsum[r] * cf + psum[r];
            rmax[r] = rm_new[r];
            o[0][r] *= cf; o[1][r] *= cf; o[2][r] *= cf; o[3][r] *= cf;
        }
        asm volatile("s_wait_dscnt 0x0" ::: "memory");     // P store->load (same wave)
        FragAB pa;
        {
            const unsigned short* lp = &lds_p[wid][lrow * 32 + lhi * 16];
            pa.u[0] = *(const ui4*)(lp);
            pa.u[1] = *(const ui4*)(lp + 8);
        }
#pragma unroll
        for (int t4 = 0; t4 < 4; t4++) {
            int vr = (t4 * 16 + lrow) * 32 + lhi * 16;
            FragAB vf;
            vf.u[0] = *(const ui4*)(vB + vr);
            vf.u[1] = *(const ui4*)(vB + vr + 8);
            o[t4] = wmma_bf16(pa, vf, o[t4]);
        }
    }
#pragma unroll
    for (int r = 0; r < 8; r++) {
        float inv = 1.f / rsum[r];
        size_t row = (size_t)(b * N_ + n0 + r + 8 * lhi);
#pragma unroll
        for (int t4 = 0; t4 < 4; t4++)
            ao[row * 512 + h * 64 + t4 * 16 + lrow] = f2bf(o[t4][r] * inv);
    }
}

// ---------------------------------------------------------------------------
// 5) Output projection: [8192,512]bf16 x [512,512] -> fp32 d_out.
// ---------------------------------------------------------------------------
__global__ void gemm_out(const unsigned short* __restrict__ ao,
                         const unsigned short* __restrict__ Wt,
                         float* __restrict__ out) {
    int wid = threadIdx.x >> 5, lane = threadIdx.x & 31;
    int tile = blockIdx.x * 4 + wid;                  // 512*32 tiles
    int rt = tile >> 5, ct = tile & 31;
    int m0 = rt * 16, j0 = ct * 16;
    int lrow = lane & 15, lhi = lane >> 4;

    const unsigned short* aB = ao + (size_t)(m0 + lrow) * 512 + lhi * 16;
    const unsigned short* bB = Wt + (size_t)(j0 + lrow) * 512 + lhi * 16;

    v8f acc = {};
#pragma unroll 4
    for (int k0 = 0; k0 < 512; k0 += 32) {
        FragAB fa, fb;
        fa.u[0] = *(const ui4*)(aB + k0);
        fa.u[1] = *(const ui4*)(aB + k0 + 8);
        fb.u[0] = *(const ui4*)(bB + k0);
        fb.u[1] = *(const ui4*)(bB + k0 + 8);
        acc = wmma_bf16(fa, fb, acc);
    }
#pragma unroll
    for (int r = 0; r < 8; r++)
        out[(size_t)(m0 + r + 8 * lhi) * 512 + j0 + lrow] = acc[r];
}

// ---------------------------------------------------------------------------
extern "C" void kernel_launch(void* const* d_in, const int* in_sizes, int n_in,
                              void* d_out, int out_size, void* d_ws, size_t ws_size,
                              hipStream_t stream) {
    const float* x     = (const float*)d_in[0];
    const float* R     = (const float*)d_in[1];
    const float* gamma = (const float*)d_in[2];
    const float* beta  = (const float*)d_in[3];
    const float* Wqkv  = (const float*)d_in[4];
    const float* Wout  = (const float*)d_in[5];
    const float* a     = (const float*)d_in[6];
    const float* c     = (const float*)d_in[7];
    float* out = (float*)d_out;

    char* w = (char*)d_ws;
    constexpr size_t SZ_XN  = (size_t)B_ * N_ * DIM_ * 2;          // 8 MB
    constexpr size_t SZ_WQ  = (size_t)1536 * 512 * 2;              // 1.5 MB
    constexpr size_t SZ_WO  = (size_t)512 * 512 * 2;               // 0.5 MB
    constexpr size_t SZ_QKV = (size_t)B_ * H_ * N_ * DH_ * 2;      // 8 MB each
    constexpr size_t SZ_AO  = (size_t)B_ * N_ * 512 * 2;           // 8 MB
    unsigned short* xn = (unsigned short*)(w);
    unsigned short* wq = (unsigned short*)(w + SZ_XN);
    unsigned short* wo = (unsigned short*)(w + SZ_XN + SZ_WQ);
    unsigned short* qd = (unsigned short*)(w + SZ_XN + SZ_WQ + SZ_WO);
    unsigned short* kd = (unsigned short*)(w + SZ_XN + SZ_WQ + SZ_WO + SZ_QKV);
    unsigned short* vT = (unsigned short*)(w + SZ_XN + SZ_WQ + SZ_WO + 2 * SZ_QKV);
    unsigned short* ao = (unsigned short*)(w + SZ_XN + SZ_WQ + SZ_WO + 3 * SZ_QKV);
    float*          de = (float*)(w + SZ_XN + SZ_WQ + SZ_WO + 3 * SZ_QKV + SZ_AO);

    ln_kernel <<<2048, 128, 0, stream>>>(x, gamma, beta, xn);
    prep_wqkv <<<(1536 * 512) / 256, 256, 0, stream>>>(Wqkv, wq);
    prep_wout <<<(512 * 512) / 256, 256, 0, stream>>>(Wout, wo);
    prep_decay<<<(B_ * H_ * T_ * T_) / 256, 256, 0, stream>>>(R, a, c, de);
    gemm_qkv  <<<(512 * 96) / 4, 128, 0, stream>>>(xn, wq, qd, kd, vT);
    attn_kernel<<<(B_ * H_ * (N_ / 16)) / 4, 128, 0, stream>>>(qd, kd, vT, de, ao);
    gemm_out  <<<(512 * 32) / 4, 128, 0, stream>>>(ao, wo, out);
}
